// HomoGAT_47828755808347
// MI455X (gfx1250) — matrix operand order repaired and checked by reading, not compiled, not run on hardware
//
#include <hip/hip_runtime.h>
#include <math.h>

#define N_NODES 50000
#define N_EDGES 800000
#define DIM_IN  184
#define HID     64
#define HEADS   4
#define OUT_DIM 2
#define HC      (HEADS * HID)       // 256
#define EE      (N_EDGES + N_NODES) // 850000 edges incl. self loops

typedef __attribute__((ext_vector_type(16))) _Float16 v16h;
typedef __attribute__((ext_vector_type(8)))  _Float16 v8h;
typedef __attribute__((ext_vector_type(8)))  float    v8f;
typedef __attribute__((ext_vector_type(4)))  int      i4;
typedef __attribute__((address_space(1))) i4* gptr_b128;  // global int4*
typedef __attribute__((address_space(3))) i4* lptr_b128;  // LDS int4*

#if __has_builtin(__builtin_amdgcn_global_load_async_to_lds_b128) && \
    __has_builtin(__builtin_amdgcn_s_wait_asynccnt)
#define HAVE_ASYNC_LDS 1
#else
#define HAVE_ASYNC_LDS 0
#endif

// ---------- order-preserving float <-> uint encoding for atomicMax ----------
__device__ __forceinline__ unsigned enc_f32(float f) {
    unsigned u = __float_as_uint(f);
    return (u & 0x80000000u) ? ~u : (u | 0x80000000u);
}
__device__ __forceinline__ float dec_f32(unsigned k) {
    unsigned u = (k & 0x80000000u) ? (k ^ 0x80000000u) : ~k;
    return __uint_as_float(u);
}
#define ENC_NEG_INF 0x007FFFFFu  // enc_f32(-inf)

// ---------- generic fill ----------
__global__ void fill_u32(unsigned* __restrict__ p, unsigned v, long long n) {
    long long i = (long long)blockIdx.x * blockDim.x + threadIdx.x;
    if (i < n) p[i] = v;
}

// ---------- weight f32 -> f16 with K-row padding ----------
__global__ void cvt_w_f16(const float* __restrict__ src, _Float16* __restrict__ dst,
                          int rows, int cols, int rowsPad) {
    int i = blockIdx.x * blockDim.x + threadIdx.x;
    if (i >= rowsPad * cols) return;
    int r = i / cols;
    dst[i] = (r < rows) ? (_Float16)src[i] : (_Float16)0.f;
}

// ---------- activation f32 [n,Kact] -> f16 [n,Krnd] (zero-pad K tail) ----------
__global__ void cvt_x_f16(const float* __restrict__ src, _Float16* __restrict__ dst,
                          int Kact, int Krnd, long long n) {
    long long i = (long long)blockIdx.x * blockDim.x + threadIdx.x;
    if (i >= n * Krnd) return;
    long long r = i / Krnd;
    int c = (int)(i - r * Krnd);
    dst[i] = (c < Kact) ? (_Float16)src[r * Kact + c] : (_Float16)0.f;
}

// ---------- WMMA GEMM: Y[nrows,Nfull](f32) = X16[nrows,Krnd](f16) * W16[Krnd,Nfull](f16) (+bias) ----------
// Block = 16 rows x Nfull cols; one wave32 per 16-wide N strip.
// A tile double-buffered in LDS, filled by GLOBAL_LOAD_ASYNC_TO_LDS_B128 (ASYNCcnt) when available.
// LDS row stride 40 halves (80B) keeps every 8-half chunk 16B-aligned and spreads banks.
__global__ __launch_bounds__(512)
void gemm_wmma_f16(const _Float16* __restrict__ X16, const _Float16* __restrict__ W16,
                   const float* __restrict__ bias, float* __restrict__ Y,
                   int Krnd, int Nfull) {
    __shared__ __align__(16) _Float16 At[2][16 * 40];

    const int tid   = threadIdx.x;
    const int lane  = tid & 31;
    const int nbase = (tid >> 5) * 16;
    const long long rowBase = (long long)blockIdx.x * 16;
    const int r0 = lane & 15;        // A: M = lane % 16
    const int kb = (lane >> 4) << 3; // A: K-half select (0 or 8)

    // loader threads 0..63 (waves 0,1 — wave-uniform predicate): one 16B chunk each
    const bool loader = (tid < 64);
    const int cr = tid >> 2;         // row 0..15
    const int ck = (tid & 3) << 3;   // half offset 0,8,16,24
    const _Float16* gsrc = X16 + (rowBase + cr) * (long long)Krnd + ck;

    auto issue = [&](int buf, int k0) {
#if HAVE_ASYNC_LDS
        __builtin_amdgcn_global_load_async_to_lds_b128(
            (gptr_b128)(gsrc + k0),
            (lptr_b128)&At[buf][cr * 40 + ck], 0, 0);
#else
        *(v8h*)&At[buf][cr * 40 + ck] = *(const v8h*)(gsrc + k0);
#endif
    };
    auto drain = [&]() {
#if HAVE_ASYNC_LDS
        __builtin_amdgcn_s_wait_asynccnt(0);
#endif
    };

    v8f acc = {};

    if (loader) { issue(0, 0); drain(); }
    __syncthreads();

    for (int k0 = 0; k0 < Krnd; k0 += 32) {
        const int cur = (k0 >> 5) & 1;
        const bool more = (k0 + 32) < Krnd;
        if (loader && more) issue(cur ^ 1, k0 + 32);   // prefetch next tile into other buffer

        // A fragment (CDNA5 16-bit A layout): lane holds row r0, K {kb..kb+7, 16+kb..16+kb+7}
        const _Float16* arow = &At[cur][r0 * 40];
        v8h lo = *(const v8h*)&arow[kb];
        v8h hi = *(const v8h*)&arow[16 + kb];
        v16h a;
#pragma unroll
        for (int e = 0; e < 8; ++e) { a[e] = lo[e]; a[8 + e] = hi[e]; }

        // B fragment (CDNA5 16-bit B layout): lane holds column N = lane%16, K {kb..kb+7, 16+kb..16+kb+7}
        v16h b;
#pragma unroll
        for (int e = 0; e < 8; ++e) {
            b[e]     = W16[(long long)(k0 + kb + e) * Nfull + nbase + r0];
            b[8 + e] = W16[(long long)(k0 + 16 + kb + e) * Nfull + nbase + r0];
        }

        acc = __builtin_amdgcn_wmma_f32_16x16x32_f16(false, a, false, b,
                                                     (short)0, acc, false, false);

        if (loader && more) drain();
        __syncthreads();
    }

    // C/D layout: element r -> (M = r + 8*(lane>=16), N = lane%16)
    const int col  = nbase + (lane & 15);
    const int rOff = (lane >> 4) << 3;
    const float bv = bias ? bias[col] : 0.0f;
#pragma unroll
    for (int r = 0; r < 8; ++r)
        Y[(rowBase + rOff + r) * Nfull + col] = acc[r] + bv;
}

// ---------- per-node attention logits: als/ald[n,h] = <hW[n,h,:], a_s/a_d[h,:]> ----------
template<int H, int C>
__global__ void attn_logits(const float* __restrict__ hW, const float* __restrict__ as,
                            const float* __restrict__ ad, float* __restrict__ als,
                            float* __restrict__ ald, int n) {
    int idx = blockIdx.x * blockDim.x + threadIdx.x;
    if (idx >= n * H) return;
    int node = idx / H, h = idx % H;
    const float4* row4 = (const float4*)(hW + (long long)node * H * C + (long long)h * C);
    const float4* as4  = (const float4*)(as + h * C);
    const float4* ad4  = (const float4*)(ad + h * C);
    float s = 0.f, d = 0.f;
#pragma unroll 4
    for (int c = 0; c < C / 4; ++c) {
        float4 v = row4[c], a = as4[c], b = ad4[c];
        s += v.x * a.x + v.y * a.y + v.z * a.z + v.w * a.w;
        d += v.x * b.x + v.y * b.y + v.z * b.z + v.w * b.w;
    }
    als[idx] = s; ald[idx] = d;
}

__device__ __forceinline__ void edge_sd(const int* __restrict__ ei, long long e, int& src, int& dst) {
    if (e < N_EDGES) { src = ei[e]; dst = ei[N_EDGES + e]; }
    else             { src = dst = (int)(e - N_EDGES); }   // self-loops
}
template<int H>
__device__ __forceinline__ float edge_logit(const float* als, const float* ald,
                                            int src, int dst, int h) {
    float v = als[(long long)src * H + h] + ald[(long long)dst * H + h];
    return (v > 0.f) ? v : 0.2f * v;   // leaky_relu 0.2
}

// ---------- segment max over dst ----------
template<int H>
__global__ void edge_max_k(const int* __restrict__ ei, const float* __restrict__ als,
                           const float* __restrict__ ald, unsigned* __restrict__ menc) {
    long long t = (long long)blockIdx.x * blockDim.x + threadIdx.x;
    if (t >= (long long)EE * H) return;
    long long e = t / H; int h = (int)(t - e * H);
    int src, dst; edge_sd(ei, e, src, dst);
    atomicMax(&menc[(long long)dst * H + h], enc_f32(edge_logit<H>(als, ald, src, dst, h)));
}

// ---------- exp(e - max) and segment sum ----------
template<int H>
__global__ void edge_sumexp_k(const int* __restrict__ ei, const float* __restrict__ als,
                              const float* __restrict__ ald, const unsigned* __restrict__ menc,
                              float* __restrict__ den, float* __restrict__ ex) {
    long long t = (long long)blockIdx.x * blockDim.x + threadIdx.x;
    if (t >= (long long)EE * H) return;
    long long e = t / H; int h = (int)(t - e * H);
    int src, dst; edge_sd(ei, e, src, dst);
    float v = edge_logit<H>(als, ald, src, dst, h);
    float xv = __expf(v - dec_f32(menc[(long long)dst * H + h]));
    ex[t] = xv;
    atomicAdd(&den[(long long)dst * H + h], xv);
}

// ---------- message scatter, 4 channels/thread: acc[dst,c..c+3] += hW[src,c..c+3] * alpha ----------
template<int H, int C>
__global__ void edge_msg4_k(const int* __restrict__ ei, const float* __restrict__ hW,
                            const float* __restrict__ ex, const float* __restrict__ den,
                            float* __restrict__ acc) {
    long long t = (long long)blockIdx.x * blockDim.x + threadIdx.x;
    const int HCl = H * C, nv = HCl >> 2;
    if (t >= (long long)EE * nv) return;
    long long e = t / nv;
    int c = (int)(t - e * nv) << 2;
    int h = c / C;
    int src, dst; edge_sd(ei, e, src, dst);
    float alpha = ex[e * H + h] / (den[(long long)dst * H + h] + 1e-16f);
    float4 sv = *(const float4*)(hW + (long long)src * HCl + c);
    float* d4 = acc + (long long)dst * HCl + c;
    atomicAdd(&d4[0], sv.x * alpha);
    atomicAdd(&d4[1], sv.y * alpha);
    atomicAdd(&d4[2], sv.z * alpha);
    atomicAdd(&d4[3], sv.w * alpha);
}

// ---------- message scatter for the 2-wide final head ----------
__global__ void edge_msg2_k(const int* __restrict__ ei, const float* __restrict__ hW2,
                            const float* __restrict__ ex, const float* __restrict__ den,
                            float* __restrict__ acc) {
    long long e = (long long)blockIdx.x * blockDim.x + threadIdx.x;
    if (e >= (long long)EE) return;
    int src, dst; edge_sd(ei, e, src, dst);
    float alpha = ex[e] / (den[dst] + 1e-16f);
    float2 sv = *(const float2*)(hW2 + (long long)src * 2);
    float* d2 = acc + (long long)dst * 2;
    atomicAdd(&d2[0], sv.x * alpha);
    atomicAdd(&d2[1], sv.y * alpha);
}

// ---------- bias + LayerNorm (+residual) + ELU, block(256) per node ----------
__global__ __launch_bounds__(HC)
void post_ln_elu(const float* __restrict__ acc, const float* __restrict__ bias,
                 const float* __restrict__ g, const float* __restrict__ bln,
                 const float* __restrict__ resid, float* __restrict__ out) {
    __shared__ float red[HC];
    const int node = blockIdx.x, c = threadIdx.x;
    const long long base = (long long)node * HC;
    float v = acc[base + c] + bias[c];
    red[c] = v; __syncthreads();
    for (int s = HC / 2; s > 0; s >>= 1) { if (c < s) red[c] += red[c + s]; __syncthreads(); }
    float mu = red[0] * (1.0f / HC);
    __syncthreads();
    float dv = v - mu;
    red[c] = dv * dv; __syncthreads();
    for (int s = HC / 2; s > 0; s >>= 1) { if (c < s) red[c] += red[c + s]; __syncthreads(); }
    float var = red[0] * (1.0f / HC);
    float t = dv * rsqrtf(var + 1e-5f) * g[c] + bln[c];
    if (resid) t += resid[base + c];
    out[base + c] = (t > 0.f) ? t : expm1f(t);
}

// ---------- final head linear (C=2) + its attention logits ----------
__global__ void lin2_k(const float* __restrict__ h, const float* __restrict__ W2,
                       const float* __restrict__ as2, const float* __restrict__ ad2,
                       float* __restrict__ hW2, float* __restrict__ als2, float* __restrict__ ald2) {
    int n = blockIdx.x * blockDim.x + threadIdx.x;
    if (n >= N_NODES) return;
    const float4* row4 = (const float4*)(h + (long long)n * HC);
    float o0 = 0.f, o1 = 0.f;
#pragma unroll 4
    for (int c = 0; c < HC / 4; ++c) {
        float4 v = row4[c];
        const float* w = W2 + c * 8;
        o0 += v.x * w[0] + v.y * w[2] + v.z * w[4] + v.w * w[6];
        o1 += v.x * w[1] + v.y * w[3] + v.z * w[5] + v.w * w[7];
    }
    hW2[n * 2] = o0; hW2[n * 2 + 1] = o1;
    als2[n] = o0 * as2[0] + o1 * as2[1];
    ald2[n] = o0 * ad2[0] + o1 * ad2[1];
}

__global__ void final_bias_k(const float* __restrict__ acc2, const float* __restrict__ b2,
                             float* __restrict__ out) {
    int i = blockIdx.x * blockDim.x + threadIdx.x;
    if (i < N_NODES * OUT_DIM) out[i] = acc2[i] + b2[i & 1];
}

// =====================================================================
extern "C" void kernel_launch(void* const* d_in, const int* in_sizes, int n_in,
                              void* d_out, int out_size, void* d_ws, size_t ws_size,
                              hipStream_t stream) {
    (void)in_sizes; (void)n_in; (void)out_size; (void)ws_size;
    const float* x   = (const float*)d_in[0];
    const int*   ei  = (const int*)  d_in[1];
    const float* Wp  = (const float*)d_in[2];
    const float* bp  = (const float*)d_in[3];
    const float* W0  = (const float*)d_in[4];
    const float* as0 = (const float*)d_in[5];
    const float* ad0 = (const float*)d_in[6];
    const float* b0  = (const float*)d_in[7];
    const float* W1  = (const float*)d_in[8];
    const float* as1 = (const float*)d_in[9];
    const float* ad1 = (const float*)d_in[10];
    const float* b1  = (const float*)d_in[11];
    const float* W2  = (const float*)d_in[12];
    const float* as2 = (const float*)d_in[13];
    const float* ad2 = (const float*)d_in[14];
    const float* b2  = (const float*)d_in[15];
    const float* lng = (const float*)d_in[16];
    const float* lnb = (const float*)d_in[17];
    float* out = (float*)d_out;

    char* ws = (char*)d_ws; size_t off = 0;
    auto alloc = [&](size_t bytes) -> char* {
        char* p = ws + off; off += (bytes + 255) & ~(size_t)255; return p;
    };
    float*    P0   = (float*)alloc((size_t)N_NODES * HID * 4);
    float*    P1   = (float*)alloc((size_t)N_NODES * HC * 4);
    float*    P2   = (float*)alloc((size_t)N_NODES * HC * 4);
    float*    P3   = (float*)alloc((size_t)N_NODES * HC * 4);
    _Float16* X16  = (_Float16*)alloc((size_t)N_NODES * HC * 2);   // padded f16 activations (reused)
    float*    als  = (float*)alloc((size_t)N_NODES * HEADS * 4);
    float*    ald  = (float*)alloc((size_t)N_NODES * HEADS * 4);
    unsigned* menc = (unsigned*)alloc((size_t)N_NODES * HEADS * 4);
    float*    den  = (float*)alloc((size_t)N_NODES * HEADS * 4);
    float*    ex   = (float*)alloc((size_t)EE * HEADS * 4);
    float*    hW2  = (float*)alloc((size_t)N_NODES * 2 * 4);
    float*    als2 = (float*)alloc((size_t)N_NODES * 4);
    float*    ald2 = (float*)alloc((size_t)N_NODES * 4);
    float*    acc2 = (float*)alloc((size_t)N_NODES * 2 * 4);
    _Float16* W16p = (_Float16*)alloc((size_t)192 * 64 * 2);
    _Float16* W16a = (_Float16*)alloc((size_t)64 * 256 * 2);
    _Float16* W16b = (_Float16*)alloc((size_t)256 * 256 * 2);

    const int rowBlocks = N_NODES / 16;   // 3125 (exact)
    auto g1 = [](long long tot) { return (unsigned)((tot + 255) / 256); };

    // weight conversion (Wp padded 184 -> 192 in K)
    cvt_w_f16<<<g1(192 * 64),  256, 0, stream>>>(Wp, W16p, DIM_IN, HID, 192);
    cvt_w_f16<<<g1(64  * 256), 256, 0, stream>>>(W0, W16a, HID, HC, HID);
    cvt_w_f16<<<g1(256 * 256), 256, 0, stream>>>(W1, W16b, HC, HC, HC);

    // input projection: h0 = x @ Wp + bp  -> P0 [N,64]
    cvt_x_f16<<<g1((long long)N_NODES * 192), 256, 0, stream>>>(x, X16, DIM_IN, 192, N_NODES);
    gemm_wmma_f16<<<rowBlocks, (HID / 16) * 32, 0, stream>>>(X16, W16p, bp, P0, 192, HID);

    // ---------------- layer 0 ----------------
    cvt_x_f16<<<g1((long long)N_NODES * HID), 256, 0, stream>>>(P0, X16, HID, HID, N_NODES);
    gemm_wmma_f16<<<rowBlocks, (HC / 16) * 32, 0, stream>>>(X16, W16a, nullptr, P1, HID, HC);
    attn_logits<HEADS, HID><<<g1((long long)N_NODES * HEADS), 256, 0, stream>>>(P1, as0, ad0, als, ald, N_NODES);
    fill_u32<<<g1((long long)N_NODES * HEADS), 256, 0, stream>>>(menc, ENC_NEG_INF, (long long)N_NODES * HEADS);
    fill_u32<<<g1((long long)N_NODES * HEADS), 256, 0, stream>>>((unsigned*)den, 0u, (long long)N_NODES * HEADS);
    fill_u32<<<g1((long long)N_NODES * HC), 256, 0, stream>>>((unsigned*)P2, 0u, (long long)N_NODES * HC);
    edge_max_k<HEADS>   <<<g1((long long)EE * HEADS), 256, 0, stream>>>(ei, als, ald, menc);
    edge_sumexp_k<HEADS><<<g1((long long)EE * HEADS), 256, 0, stream>>>(ei, als, ald, menc, den, ex);
    edge_msg4_k<HEADS, HID><<<g1((long long)EE * (HC / 4)), 256, 0, stream>>>(ei, P1, ex, den, P2);
    post_ln_elu<<<N_NODES, HC, 0, stream>>>(P2, b0, lng, lnb, nullptr, P3);   // h1 -> P3

    // ---------------- layer 1 ----------------
    cvt_x_f16<<<g1((long long)N_NODES * HC), 256, 0, stream>>>(P3, X16, HC, HC, N_NODES);
    gemm_wmma_f16<<<rowBlocks, (HC / 16) * 32, 0, stream>>>(X16, W16b, nullptr, P1, HC, HC);
    attn_logits<HEADS, HID><<<g1((long long)N_NODES * HEADS), 256, 0, stream>>>(P1, as1, ad1, als, ald, N_NODES);
    fill_u32<<<g1((long long)N_NODES * HEADS), 256, 0, stream>>>(menc, ENC_NEG_INF, (long long)N_NODES * HEADS);
    fill_u32<<<g1((long long)N_NODES * HEADS), 256, 0, stream>>>((unsigned*)den, 0u, (long long)N_NODES * HEADS);
    fill_u32<<<g1((long long)N_NODES * HC), 256, 0, stream>>>((unsigned*)P2, 0u, (long long)N_NODES * HC);
    edge_max_k<HEADS>   <<<g1((long long)EE * HEADS), 256, 0, stream>>>(ei, als, ald, menc);
    edge_sumexp_k<HEADS><<<g1((long long)EE * HEADS), 256, 0, stream>>>(ei, als, ald, menc, den, ex);
    edge_msg4_k<HEADS, HID><<<g1((long long)EE * (HC / 4)), 256, 0, stream>>>(ei, P1, ex, den, P2);
    post_ln_elu<<<N_NODES, HC, 0, stream>>>(P2, b1, lng, lnb, P3, P1);        // h2 -> P1 (resid = P3)

    // ---------------- layer 2 (heads=1, C=2) ----------------
    lin2_k<<<g1(N_NODES), 256, 0, stream>>>(P1, W2, as2, ad2, hW2, als2, ald2);
    fill_u32<<<g1(N_NODES), 256, 0, stream>>>(menc, ENC_NEG_INF, (long long)N_NODES);
    fill_u32<<<g1(N_NODES), 256, 0, stream>>>((unsigned*)den, 0u, (long long)N_NODES);
    fill_u32<<<g1((long long)N_NODES * OUT_DIM), 256, 0, stream>>>((unsigned*)acc2, 0u, (long long)N_NODES * OUT_DIM);
    edge_max_k<1>   <<<g1((long long)EE), 256, 0, stream>>>(ei, als2, ald2, menc);
    edge_sumexp_k<1><<<g1((long long)EE), 256, 0, stream>>>(ei, als2, ald2, menc, den, ex);
    edge_msg2_k  <<<g1((long long)EE), 256, 0, stream>>>(ei, hW2, ex, den, acc2);
    final_bias_k<<<g1((long long)N_NODES * OUT_DIM), 256, 0, stream>>>(acc2, b2, out);
}
